// QuantumCircuitGNN_80985903333955
// MI455X (gfx1250) — compile-verified, matrix operand order checked by reading, not run on hardware
//
#include <hip/hip_runtime.h>
#include <cstdint>

#define HDIM   512
#define HEADS  4
#define DHEAD  128
#define GRAPHS 64
#define TB     256

typedef __attribute__((ext_vector_type(16))) __bf16 v16bf;
typedef __attribute__((ext_vector_type(8)))  float  v8f;
typedef __attribute__((ext_vector_type(4)))  unsigned int v4u;
typedef __attribute__((ext_vector_type(8)))  int v8i;
typedef __attribute__((ext_vector_type(4)))  int v4i;

static __device__ __forceinline__ unsigned short f2bf(float f) {
  unsigned int u = __float_as_uint(f);
  u += 0x7FFFu + ((u >> 16) & 1u);           // round-to-nearest-even
  return (unsigned short)(u >> 16);
}

// Build a 16-element bf16 fragment from two 16-byte chunks (A/B operand lane layout:
// lanes 0-15 take K-halves {0..7,16..23}, lanes 16-31 take {8..15,24..31} of each 32-K step).
static __device__ __forceinline__ v16bf make_frag(const unsigned short* p0,
                                                  const unsigned short* p1) {
  union { uint4 q[2]; v16bf v; } u;
  u.q[0] = *(const uint4*)p0;
  u.q[1] = *(const uint4*)p1;
  return u.v;
}

// ---------------- conversion / init kernels ----------------

__global__ void pad_x_bf16(const float* __restrict__ x, unsigned short* __restrict__ out,
                           int N, int Fin) {
  int idx = blockIdx.x * blockDim.x + threadIdx.x;
  if (idx >= N * 32) return;
  int n = idx >> 5, k = idx & 31;
  out[idx] = (k < Fin) ? f2bf(x[n * Fin + k]) : (unsigned short)0;
}

// W is [K, 512] row-major; WT is [512, Kp] row-major (transposed, zero-padded K)
__global__ void transpose_convert_bf16(const float* __restrict__ W,
                                       unsigned short* __restrict__ WT,
                                       int K, int Kp) {
  int idx = blockIdx.x * blockDim.x + threadIdx.x;
  if (idx >= HDIM * Kp) return;
  int n = idx / Kp, k = idx % Kp;
  WT[idx] = (k < K) ? f2bf(W[(size_t)k * HDIM + n]) : (unsigned short)0;
}

__global__ void fill_f32(float* __restrict__ p, float v, int n) {
  int idx = blockIdx.x * blockDim.x + threadIdx.x;
  if (idx < n) p[idx] = v;
}

__global__ void relu_to_bf16(const float* __restrict__ in, unsigned short* __restrict__ out,
                             int n) {
  int idx = blockIdx.x * blockDim.x + threadIdx.x;
  if (idx >= n) return;
  float v = in[idx];
  out[idx] = f2bf(v > 0.f ? v : 0.f);
}

// ---------------- WMMA GEMM: C[M,512] = A[M,Kp](bf16) @ WT^T + bias ----------------
// One block per 16-row tile. Wave 0 TDM-loads the 16xKp A tile into LDS
// (tensor_load_to_lds + s_wait_tensorcnt), then the 8 waves each compute a
// 16x64 output strip: A fragments from LDS (ds_load_b128), B from L2-resident
// transposed weights, 4 WMMA accumulators per wave. Branch-free inner loop.

static __device__ __forceinline__ void tdm_load_tile(const unsigned short* gsrc,
                                                     unsigned lds_byte_addr,
                                                     int Kp) {
  unsigned long long ga = (unsigned long long)(uintptr_t)gsrc;
  unsigned dim0 = (unsigned)Kp;      // tile/tensor width in 2-byte elements
  unsigned dim1 = 16u;               // rows
  v4u g0;
  g0.x = 1u;                                                  // count=1, user mode
  g0.y = lds_byte_addr;                                       // lds_addr
  g0.z = (unsigned)(ga & 0xFFFFFFFFull);                      // global_addr[31:0]
  g0.w = (unsigned)((ga >> 32) & 0x01FFFFFFull) | (2u << 30); // addr[56:32] | type=2
  v8i g1;
  g1[0] = (int)(1u << 16);                                    // data_size=1 -> 2 bytes
  g1[1] = (int)((dim0 & 0xFFFFu) << 16);                      // tensor_dim0[15:0] @ bit48
  g1[2] = (int)((dim0 >> 16) | ((dim1 & 0xFFFFu) << 16));     // dim0 hi | tensor_dim1 lo
  g1[3] = (int)((dim1 >> 16) | ((dim0 & 0xFFFFu) << 16));     // dim1 hi | tile_dim0
  g1[4] = (int)(dim1 & 0xFFFFu);                              // tile_dim1=16, tile_dim2=0
  g1[5] = (int)dim0;                                          // tensor_dim0_stride[31:0]
  g1[6] = 0;                                                  // stride0 hi | stride1 lo
  g1[7] = 0;                                                  // stride1 hi
  v4i gz = {0, 0, 0, 0};
#if __has_include(<hip/amd_detail/amd_gfx1250_TDM.h>)
  v8i gz8 = {0, 0, 0, 0, 0, 0, 0, 0};
  __builtin_amdgcn_tensor_load_to_lds(g0, g1, gz, gz, gz8, 0);
#else
  __builtin_amdgcn_tensor_load_to_lds(g0, g1, gz, gz, 0);
#endif
  __builtin_amdgcn_s_wait_tensorcnt(0);
}

__global__ void __launch_bounds__(TB)
gemm_bf16_wmma(const unsigned short* __restrict__ A,
               const unsigned short* __restrict__ WT,
               const float* __restrict__ bias,
               float* __restrict__ C,
               int Mtiles, int Kp) {
  __shared__ unsigned short ldsA[16 * HDIM];   // up to 16 KB

  int rt = blockIdx.x;
  if (rt >= Mtiles) return;
  int lane = threadIdx.x & 31;
  int cg = threadIdx.x >> 5;     // column group of 64 (0..7)

  if (threadIdx.x < 32) {
    tdm_load_tile(A + (size_t)rt * 16 * Kp,
                  (unsigned)(uintptr_t)(void*)ldsA, Kp);
  }
  __syncthreads();

  int lrow  = lane & 15;
  int lhalf = lane >> 4;
  const unsigned short* arow = ldsA + (size_t)lrow * Kp;
  const unsigned short* bbase = WT + (size_t)(cg * 64 + lrow) * Kp;

  v8f zero = {0.f, 0.f, 0.f, 0.f, 0.f, 0.f, 0.f, 0.f};
  v8f acc[4];
#pragma unroll
  for (int t = 0; t < 4; ++t) acc[t] = zero;

  for (int kb = 0; kb < Kp; kb += 32) {
    int k0 = kb + lhalf * 8;
    int k1 = k0 + 16;
    v16bf a = make_frag(arow + k0, arow + k1);     // ds_load_b128 x2
#pragma unroll
    for (int t = 0; t < 4; ++t) {
      const unsigned short* brow = bbase + (size_t)t * 16 * Kp;
      // unconditional speculative prefetch of next K-chunk (dropped if OOB)
      __builtin_prefetch(brow + k0 + 32, 0, 3);    // global_prefetch_b8
      v16bf b = make_frag(brow + k0, brow + k1);
      acc[t] = __builtin_amdgcn_wmma_f32_16x16x32_bf16(
          false, a, false, b, (short)0, acc[t], false, false);
    }
  }

#pragma unroll
  for (int t = 0; t < 4; ++t) {
    int col = cg * 64 + t * 16 + lrow;
    float bv = bias[col];
    float* cc = C + (size_t)(rt * 16 + 8 * lhalf) * HDIM + col;
#pragma unroll
    for (int r = 0; r < 8; ++r)
      cc[(size_t)r * HDIM] = acc[t][r] + bv;
  }
}

// ---------------- edge phase ----------------

// one wave per edge; lanes 0-7 -> head0, 8-15 -> head1, ...
__global__ void edge_logits_kernel(const float* __restrict__ Q, const float* __restrict__ Kf,
                                   const int* __restrict__ src, const int* __restrict__ dst,
                                   float* __restrict__ alpha, int E) {
  int gid = blockIdx.x * blockDim.x + threadIdx.x;
  int e = gid >> 5, lane = gid & 31;
  if (e >= E) return;
  int s = src[e], d = dst[e];
  const float4* qp = (const float4*)(Q  + (size_t)d * HDIM + lane * 16);
  const float4* kp = (const float4*)(Kf + (size_t)s * HDIM + lane * 16);
  float part = 0.f;
#pragma unroll
  for (int i = 0; i < 4; ++i) {
    float4 q4 = qp[i], k4 = kp[i];
    part += q4.x * k4.x + q4.y * k4.y + q4.z * k4.z + q4.w * k4.w;
  }
  part += __shfl_xor(part, 1, 32);
  part += __shfl_xor(part, 2, 32);
  part += __shfl_xor(part, 4, 32);
  if ((lane & 7) == 0)
    alpha[(size_t)e * HEADS + (lane >> 3)] = part * 0.08838834764831845f; // 1/sqrt(128)
}

// sign-split float atomic max -> native int atomics (amax initialized to -inf)
static __device__ __forceinline__ void atomicMaxFloat(float* addr, float v) {
  if (v >= 0.f) atomicMax((int*)addr, __float_as_int(v));
  else          atomicMin((unsigned int*)addr, __float_as_uint(v));
}

__global__ void edge_max_kernel(const float* __restrict__ alpha, const int* __restrict__ dst,
                                float* __restrict__ amax, int E) {
  int idx = blockIdx.x * blockDim.x + threadIdx.x;
  if (idx >= E * HEADS) return;
  int e = idx / HEADS, h = idx % HEADS;
  atomicMaxFloat(&amax[(size_t)dst[e] * HEADS + h], alpha[idx]);
}

__global__ void edge_exp_kernel(float* __restrict__ alpha, const int* __restrict__ dst,
                                const float* __restrict__ amax, float* __restrict__ den,
                                int E) {
  int idx = blockIdx.x * blockDim.x + threadIdx.x;
  if (idx >= E * HEADS) return;
  int e = idx / HEADS, h = idx % HEADS;
  int d = dst[e];
  float ex = __expf(alpha[idx] - amax[(size_t)d * HEADS + h]);
  alpha[idx] = ex;
  atomicAdd(&den[(size_t)d * HEADS + h], ex);
}

// thread per (edge, 4-float chunk): 128 threads/edge
__global__ void edge_agg_kernel(const float* __restrict__ ex, const float* __restrict__ den,
                                const float* __restrict__ V, const int* __restrict__ src,
                                const int* __restrict__ dst, float* __restrict__ agg, int E) {
  int idx = blockIdx.x * blockDim.x + threadIdx.x;
  if (idx >= E * 128) return;
  int e = idx >> 7, c = idx & 127;
  int h = c >> 5;                      // 32 chunks (128 floats) per head
  int s = src[e], d = dst[e];
  float w = ex[(size_t)e * HEADS + h] / den[(size_t)d * HEADS + h];
  float4 v4 = *(const float4*)(V + (size_t)s * HDIM + c * 4);
  float* a = agg + (size_t)d * HDIM + c * 4;
  atomicAdd(a + 0, w * v4.x);
  atomicAdd(a + 1, w * v4.y);
  atomicAdd(a + 2, w * v4.z);
  atomicAdd(a + 3, w * v4.w);
}

// ---------------- pooling + MLP ----------------

__global__ void pool_count_kernel(const int* __restrict__ batch, float* __restrict__ cnt, int N) {
  int idx = blockIdx.x * blockDim.x + threadIdx.x;
  if (idx < N) atomicAdd(&cnt[batch[idx]], 1.0f);
}

__global__ void pool_sum_kernel(const float* __restrict__ agg, const int* __restrict__ batch,
                                float* __restrict__ gp, int N) {
  int idx = blockIdx.x * blockDim.x + threadIdx.x;
  if (idx >= N * 128) return;
  int n = idx >> 7, c = idx & 127;
  float4 v4 = *(const float4*)(agg + (size_t)n * HDIM + c * 4);
  float* g = gp + (size_t)batch[n] * HDIM + c * 4;
  atomicAdd(g + 0, fmaxf(v4.x, 0.f));   // relu of layer-2 output fused here
  atomicAdd(g + 1, fmaxf(v4.y, 0.f));
  atomicAdd(g + 2, fmaxf(v4.z, 0.f));
  atomicAdd(g + 3, fmaxf(v4.w, 0.f));
}

__global__ void mlp1_kernel(const float* __restrict__ gp, const float* __restrict__ cnt,
                            const float* __restrict__ Wd1, const float* __restrict__ bd1,
                            float* __restrict__ hid) {
  int idx = blockIdx.x * blockDim.x + threadIdx.x;
  if (idx >= GRAPHS * 128) return;
  int g = idx >> 7, j = idx & 127;
  float invc = 1.0f / cnt[g];
  float sum = bd1[j];
  const float* gr = gp + (size_t)g * HDIM;
  for (int f = 0; f < HDIM; ++f)
    sum += gr[f] * invc * Wd1[(size_t)f * 128 + j];
  hid[idx] = fmaxf(sum, 0.f);
}

__global__ void mlp2_kernel(const float* __restrict__ hid, const float* __restrict__ Wd2,
                            const float* __restrict__ bd2, float* __restrict__ out) {
  int g = blockIdx.x * blockDim.x + threadIdx.x;
  if (g >= GRAPHS) return;
  float sum = bd2[0];
  const float* hr = hid + (size_t)g * 128;
  for (int j = 0; j < 128; ++j) sum += hr[j] * Wd2[j];
  out[g] = sum;
}

// ---------------- host ----------------

static inline int cdiv(long long a, int b) { return (int)((a + b - 1) / b); }

extern "C" void kernel_launch(void* const* d_in, const int* in_sizes, int n_in,
                              void* d_out, int out_size, void* d_ws, size_t ws_size,
                              hipStream_t stream) {
  const float* x     = (const float*)d_in[0];
  const int*   eidx  = (const int*)d_in[1];
  const int*   batch = (const int*)d_in[2];
  const float* W1[4] = {(const float*)d_in[3], (const float*)d_in[5],
                        (const float*)d_in[7], (const float*)d_in[9]};
  const float* b1[4] = {(const float*)d_in[4], (const float*)d_in[6],
                        (const float*)d_in[8], (const float*)d_in[10]};
  const float* W2[4] = {(const float*)d_in[11], (const float*)d_in[13],
                        (const float*)d_in[15], (const float*)d_in[17]};
  const float* b2[4] = {(const float*)d_in[12], (const float*)d_in[14],
                        (const float*)d_in[16], (const float*)d_in[18]};
  const float* Wd1 = (const float*)d_in[19];
  const float* bd1 = (const float*)d_in[20];
  const float* Wd2 = (const float*)d_in[21];
  const float* bd2 = (const float*)d_in[22];

  const int N = in_sizes[2];        // 30000 (multiple of 16)
  const int E = in_sizes[1] / 2;    // 300000
  const int Fin = in_sizes[0] / N;  // 5
  const int* src = eidx;
  const int* dst = eidx + E;

  char* p = (char*)d_ws;
  auto alloc = [&](size_t bytes) -> void* {
    void* r = (void*)p;
    p += (bytes + 255) & ~(size_t)255;
    return r;
  };
  unsigned short* Apad = (unsigned short*)alloc((size_t)N * 32 * 2);
  unsigned short* Hbf  = (unsigned short*)alloc((size_t)N * HDIM * 2);
  unsigned short* WT   = (unsigned short*)alloc((size_t)4 * HDIM * HDIM * 2);
  float* Qb    = (float*)alloc((size_t)N * HDIM * 4);
  float* Kb    = (float*)alloc((size_t)N * HDIM * 4);
  float* Vb    = (float*)alloc((size_t)N * HDIM * 4);
  float* AGG   = (float*)alloc((size_t)N * HDIM * 4);
  float* ALPHA = (float*)alloc((size_t)E * HEADS * 4);
  float* AMAX  = (float*)alloc((size_t)N * HEADS * 4);
  float* DEN   = (float*)alloc((size_t)N * HEADS * 4);
  float* GP    = (float*)alloc((size_t)GRAPHS * HDIM * 4);
  float* CNT   = (float*)alloc((size_t)GRAPHS * 4);
  float* HID   = (float*)alloc((size_t)GRAPHS * 128 * 4);

  const int Mtiles = N / 16;

  auto run_layer = [&](const unsigned short* Abf, int Kp, int Korig,
                       const float* const* W, const float* const* b) {
    size_t wtStride = (size_t)HDIM * Kp;
    float* dests[4] = {Qb, Kb, Vb, AGG};   // Ws projection seeds AGG (skip connection)
    for (int m = 0; m < 4; ++m) {
      transpose_convert_bf16<<<cdiv((long long)HDIM * Kp, TB), TB, 0, stream>>>(
          W[m], WT + m * wtStride, Korig, Kp);
      gemm_bf16_wmma<<<Mtiles, TB, 0, stream>>>(
          Abf, WT + m * wtStride, b[m], dests[m], Mtiles, Kp);
    }
    fill_f32<<<cdiv((long long)N * HEADS, TB), TB, 0, stream>>>(
        AMAX, -__builtin_huge_valf(), N * HEADS);
    fill_f32<<<cdiv((long long)N * HEADS, TB), TB, 0, stream>>>(DEN, 0.f, N * HEADS);
    edge_logits_kernel<<<cdiv((long long)E * 32, TB), TB, 0, stream>>>(
        Qb, Kb, src, dst, ALPHA, E);
    edge_max_kernel<<<cdiv((long long)E * HEADS, TB), TB, 0, stream>>>(ALPHA, dst, AMAX, E);
    edge_exp_kernel<<<cdiv((long long)E * HEADS, TB), TB, 0, stream>>>(ALPHA, dst, AMAX, DEN, E);
    edge_agg_kernel<<<cdiv((long long)E * 128, TB), TB, 0, stream>>>(
        ALPHA, DEN, Vb, src, dst, AGG, E);
  };

  // layer 1 (K padded 5 -> 32)
  pad_x_bf16<<<cdiv((long long)N * 32, TB), TB, 0, stream>>>(x, Apad, N, Fin);
  run_layer(Apad, 32, Fin, W1, b1);
  relu_to_bf16<<<cdiv((long long)N * HDIM, TB), TB, 0, stream>>>(AGG, Hbf, N * HDIM);

  // layer 2 (K = 512)
  run_layer(Hbf, HDIM, HDIM, W2, b2);

  // global mean pool (relu fused) + MLP head
  fill_f32<<<cdiv((long long)GRAPHS * HDIM, TB), TB, 0, stream>>>(GP, 0.f, GRAPHS * HDIM);
  fill_f32<<<1, GRAPHS, 0, stream>>>(CNT, 0.f, GRAPHS);
  pool_count_kernel<<<cdiv(N, TB), TB, 0, stream>>>(batch, CNT, N);
  pool_sum_kernel<<<cdiv((long long)N * 128, TB), TB, 0, stream>>>(AGG, batch, GP, N);
  mlp1_kernel<<<cdiv(GRAPHS * 128, TB), TB, 0, stream>>>(GP, CNT, Wd1, bd1, HID);
  mlp2_kernel<<<1, GRAPHS, 0, stream>>>(HID, Wd2, bd2, (float*)d_out);
}
